// EuclideanContrastiveLoss_73461120630910
// MI455X (gfx1250) — matrix operand order repaired and checked by reading, not compiled, not run on hardware
//
#include <hip/hip_runtime.h>
#include <hip/hip_bf16.h>
#include <hip/hip_fp16.h>

// ---------------------------------------------------------------------------
// EuclideanContrastiveLoss on MI455X (gfx1250, wave32, WMMA)
//
// Roofline: Gram matrix = 8192^2 * 512 * 2 = 68.7 GFLOP on 16 MB of data
// (fits in 192 MB L2) -> compute bound -> v_wmma_f32_16x16x32_bf16 with fp32
// accumulation. logits in [-28.6, 0] so exp() never over/underflows in fp32:
// single fused pass, no max-shift needed.
//
// Data movement: double-buffered GLOBAL_LOAD_ASYNC_TO_LDS_B128 (ASYNCcnt)
// so the async engine stages chunk k+1 while WMMAs consume chunk k.
// ---------------------------------------------------------------------------

typedef __attribute__((ext_vector_type(16))) __bf16 v16bf;
typedef __attribute__((ext_vector_type(8)))  float  v8f;
typedef int v4i_ __attribute__((vector_size(16)));   // matches builtin proto

#define BM 128
#define BN 128
#define BK 32
#define LDT 40   // LDS row stride in bf16 units (80 B: 16B-aligned, staggers banks)

union V16 { uint4 u[2]; v16bf v; };

#if __has_builtin(__builtin_amdgcn_global_load_async_to_lds_b128)
#define USE_ASYNC_LDS 1
#endif

__device__ __forceinline__ void copy16_to_lds(__hip_bfloat16* dst,
                                              const __hip_bfloat16* src) {
#ifdef USE_ASYNC_LDS
  // global -> LDS via the async path (no VGPR round-trip, tracked by ASYNCcnt).
  // Builtin signature (from hipcc diagnostic): (v4i AS1*, v4i AS3*, imm, imm).
  typedef __attribute__((address_space(1))) v4i_* gp_t;
  typedef __attribute__((address_space(3))) v4i_* lp_t;
  gp_t g = (gp_t)(__attribute__((address_space(1))) void*)(void*)
               const_cast<__hip_bfloat16*>(src);
  lp_t l = (lp_t)(__attribute__((address_space(3))) void*)(void*)dst;
  __builtin_amdgcn_global_load_async_to_lds_b128(g, l, 0, 0);
#else
  *(uint4*)dst = *(const uint4*)src;
#endif
}

__device__ __forceinline__ void async_wait() {
#ifdef USE_ASYNC_LDS
#if __has_builtin(__builtin_amdgcn_s_wait_asynccnt)
  __builtin_amdgcn_s_wait_asynccnt(0);
#else
  asm volatile("s_wait_asynccnt 0x0" ::: "memory");
#endif
#endif
}

// ---- Kernel 1: L2-normalize rows to bf16, zero per-row accumulators -------
__global__ __launch_bounds__(256) void normalize_rows_kernel(
    const float* __restrict__ tokens, __hip_bfloat16* __restrict__ Tn,
    float* __restrict__ S, float* __restrict__ PL, float* __restrict__ C,
    int N, int D)
{
  __shared__ float red[256];
  const int row = blockIdx.x;
  const int tid = threadIdx.x;
  const float* rp = tokens + (size_t)row * D;
  float ss = 0.f;
  for (int j = tid; j < D; j += 256) { float x = rp[j]; ss += x * x; }
  red[tid] = ss;
  __syncthreads();
  for (int s = 128; s > 0; s >>= 1) {
    if (tid < s) red[tid] += red[tid + s];
    __syncthreads();
  }
  const float scale = 1.0f / fmaxf(sqrtf(red[0]), 1e-12f);
  __hip_bfloat16* op = Tn + (size_t)row * D;
  for (int j = tid; j < D; j += 256) op[j] = __float2bfloat16(rp[j] * scale);
  if (tid == 0) { S[row] = 0.f; PL[row] = 0.f; C[row] = 0.f; }
}

// ---- Kernel 2: fused Gram-matrix + softmax-statistics pass ----------------
// Block: 128x128 output tile, 8 waves (4 along M x 2 along N),
// each wave owns a 32x64 sub-tile = 2x4 accumulators of 16x16.
__global__ __launch_bounds__(256) void gram_loss_kernel(
    const __hip_bfloat16* __restrict__ Tn, const int* __restrict__ labels,
    const float* __restrict__ temp,
    float* __restrict__ S, float* __restrict__ PL, float* __restrict__ C,
    int N, int D)
{
  __shared__ __align__(16) __hip_bfloat16 As[2][BM * LDT];
  __shared__ __align__(16) __hip_bfloat16 Bs[2][BN * LDT];
  __shared__ int labi[BM];
  __shared__ int labj[BN];

  const int ibase = blockIdx.x * BM;   // output rows
  const int jbase = blockIdx.y * BN;   // output cols
  const int tid   = threadIdx.x;
  const int lane  = tid & 31;
  const int wid   = tid >> 5;
  const int wm    = wid & 3;           // wave M index (0..3) -> rows wm*32..+32
  const int wn    = wid >> 2;          // wave N index (0..1) -> cols wn*64..+64
  const int half  = lane >> 4;         // lane group (0: lanes 0-15, 1: 16-31)
  const int l     = lane & 15;

  if (tid < BM) labi[tid] = labels[ibase + tid];
  else          labj[tid - BM] = labels[jbase + (tid - BM)];

  // Per-thread staging geometry: 512 16B chunks per matrix / 256 threads = 2.
  // Chunk c covers LDS row (srow + 64*c), bf16 col scol (8 bf16 = 16 B).
  const int srow = tid >> 2;
  const int scol = (tid & 3) * 8;
  const __hip_bfloat16* gA0 = Tn + (size_t)(ibase + srow) * D + scol;
  const __hip_bfloat16* gA1 = Tn + (size_t)(ibase + srow + 64) * D + scol;
  const __hip_bfloat16* gB0 = Tn + (size_t)(jbase + srow) * D + scol;
  const __hip_bfloat16* gB1 = Tn + (size_t)(jbase + srow + 64) * D + scol;
  const int l0 = srow * LDT + scol;
  const int l1 = (srow + 64) * LDT + scol;

  v8f acc[2][4] = {};

  // Prologue: stage chunk 0 into buffer 0.
  copy16_to_lds(&As[0][l0], gA0);
  copy16_to_lds(&As[0][l1], gA1);
  copy16_to_lds(&Bs[0][l0], gB0);
  copy16_to_lds(&Bs[0][l1], gB1);
  async_wait();
  __syncthreads();

  const int KITER = D / BK;
  for (int it = 0; it < KITER; ++it) {
    const int cur = it & 1;

    // Kick off async staging of the next chunk into the other buffer while
    // the matrix pipes chew on the current one.
    if (it + 1 < KITER) {
      const int kb = (it + 1) * BK;
      const int nxt = cur ^ 1;
      copy16_to_lds(&As[nxt][l0], gA0 + kb);
      copy16_to_lds(&As[nxt][l1], gA1 + kb);
      copy16_to_lds(&Bs[nxt][l0], gB0 + kb);
      copy16_to_lds(&Bs[nxt][l1], gB1 + kb);
    }

    // Fragment loads (ds_load_b128), per ISA 16-bit A/B layouts:
    // A (16x32): lanes 0-15 VGPR0-3 = K0..7, VGPR4-7 = K16..23;
    //            lanes 16-31 hold K8..15 / K24..31.
    v16bf a[2], b[4];
#pragma unroll
    for (int mi = 0; mi < 2; ++mi) {
      const char* rp = (const char*)&As[cur][(wm * 32 + mi * 16 + l) * LDT];
      V16 t;
      t.u[0] = *(const uint4*)(rp + half * 16);
      t.u[1] = *(const uint4*)(rp + 32 + half * 16);
      a[mi] = t.v;
    }
    // B (32x16): lanes 0-15 hold K0..7 / K16..23, lanes 16-31 hold K8..15 / K24..31 (col = lane%16)
#pragma unroll
    for (int ni = 0; ni < 4; ++ni) {
      const char* rp = (const char*)&Bs[cur][(wn * 64 + ni * 16 + l) * LDT];
      V16 t;
      t.u[0] = *(const uint4*)(rp + half * 16);
      t.u[1] = *(const uint4*)(rp + 32 + half * 16);
      b[ni] = t.v;
    }

#pragma unroll
    for (int mi = 0; mi < 2; ++mi)
#pragma unroll
      for (int ni = 0; ni < 4; ++ni)
        acc[mi][ni] = __builtin_amdgcn_wmma_f32_16x16x32_bf16(
            false, a[mi], false, b[ni], (short)0, acc[mi][ni], false, false);

    // My async writes into the next buffer are done; barrier makes them
    // (and everyone's reads of the current buffer) visible block-wide.
    async_wait();
    __syncthreads();
  }

  // ---- Fused epilogue: sim -> dist -> logits -> per-row partial stats -----
  const float invT = 1.0f / temp[0];
  const bool diagblk = (ibase == jbase);

  int labcol[4], gcol[4];
#pragma unroll
  for (int ni = 0; ni < 4; ++ni) {
    const int cl = wn * 64 + ni * 16 + l;
    labcol[ni] = labj[cl];
    gcol[ni] = jbase + cl;
  }

  // C/D layout: VGPR r holds (M=r, N=lane) for lanes 0-15 and (M=r+8, N=lane-16)
  // for lanes 16-31 -> row = base + half*8 + r, col = l.
#pragma unroll
  for (int mi = 0; mi < 2; ++mi) {
#pragma unroll
    for (int r = 0; r < 8; ++r) {
      const int rl = wm * 32 + mi * 16 + half * 8 + r;
      const int grow = ibase + rl;
      const int lrow = labi[rl];
      float es = 0.f, pls = 0.f, cs = 0.f;
#pragma unroll
      for (int ni = 0; ni < 4; ++ni) {
        const float s = acc[mi][ni][r];
        const float d2 = fmaxf(2.0f - 2.0f * s, 0.0f);
        const float logit = -sqrtf(d2) * invT;     // in [-2/T, 0]
        const bool diag = diagblk && (grow == gcol[ni]);
        const float e = diag ? 0.0f : __expf(logit);
        const bool pos = (!diag) && (lrow == labcol[ni]);
        es  += e;
        pls += pos ? logit : 0.0f;
        cs  += pos ? 1.0f : 0.0f;
      }
      // reduce the 16 columns held across each half-wave
#pragma unroll
      for (int off = 8; off > 0; off >>= 1) {
        es  += __shfl_xor(es,  off, 16);
        pls += __shfl_xor(pls, off, 16);
        cs  += __shfl_xor(cs,  off, 16);
      }
      if (l == 0) {   // lanes 0 and 16 each own one row's partial
        atomicAdd(&S[grow],  es);
        atomicAdd(&PL[grow], pls);
        atomicAdd(&C[grow],  cs);
      }
    }
  }
}

// ---- Kernel 3: reduce per-row stats to the scalar loss --------------------
__global__ __launch_bounds__(256) void finalize_kernel(
    const float* __restrict__ S, const float* __restrict__ PL,
    const float* __restrict__ C, float* __restrict__ out, int N)
{
  __shared__ float sn[256], sd[256];
  const int tid = threadIdx.x;
  float num = 0.f, den = 0.f;
  for (int i = tid; i < N; i += 256) {
    const float c = C[i];
    num += c * logf(S[i]) - PL[i];   // C_i * logsumexp_i - sum_pos(logit)
    den += c;
  }
  sn[tid] = num; sd[tid] = den;
  __syncthreads();
  for (int s = 128; s > 0; s >>= 1) {
    if (tid < s) { sn[tid] += sn[tid + s]; sd[tid] += sd[tid + s]; }
    __syncthreads();
  }
  if (tid == 0) out[0] = sn[0] / sd[0];
}

// ---------------------------------------------------------------------------
extern "C" void kernel_launch(void* const* d_in, const int* in_sizes, int n_in,
                              void* d_out, int out_size, void* d_ws, size_t ws_size,
                              hipStream_t stream) {
  const float* tokens = (const float*)d_in[0];
  const int*   labels = (const int*)d_in[1];
  const float* temp   = (const float*)d_in[2];
  float*       out    = (float*)d_out;

  const int N = in_sizes[1];           // 8192
  const int D = in_sizes[0] / N;       // 512

  // Workspace layout: [bf16 normalized matrix | S | PL | C]  (~8.1 MB)
  char* ws = (char*)d_ws;
  __hip_bfloat16* Tn = (__hip_bfloat16*)ws;
  float* S  = (float*)(ws + (size_t)N * D * sizeof(__hip_bfloat16));
  float* PL = S + N;
  float* C  = PL + N;

  normalize_rows_kernel<<<N, 256, 0, stream>>>(tokens, Tn, S, PL, C, N, D);

  dim3 grid(N / BM, N / BN);
  gram_loss_kernel<<<grid, 256, 0, stream>>>(Tn, labels, temp, S, PL, C, N, D);

  finalize_kernel<<<1, 256, 0, stream>>>(S, PL, C, out, N);
}
